// Decoder_1030792151056
// MI455X (gfx1250) — compile-verified, matrix operand order checked by reading, not run on hardware
//
#include <hip/hip_runtime.h>
#include <hip/hip_bf16.h>

// Problem constants (match reference):
// B=32, F=64, N=2048, E=512, L=32
#define BB 32
#define FF 64
#define NN 2048
#define EE 512
#define LL 32
#define K_IH   (EE + 2*NN)        // 4608
#define K_CAT  (EE + 2*NN + NN)   // 6656  ([emb | ctx | h])
#define N4     (4*NN)             // 8192
#define N3     (3*NN)             // 6144
#define N2     (2*NN)             // 4096

typedef __bf16 v16bf __attribute__((ext_vector_type(16)));
typedef float  v8f   __attribute__((ext_vector_type(8)));
typedef unsigned short v8u __attribute__((ext_vector_type(8)));

union FragAB { v16bf bf; v8u u[2]; };

__device__ __forceinline__ unsigned short f2bf(float f) {
    unsigned u = __float_as_uint(f);
    u += 0x7fffu + ((u >> 16) & 1u);        // round-to-nearest-even
    return (unsigned short)(u >> 16);
}
__device__ __forceinline__ float bf2f(unsigned short h) {
    return __uint_as_float(((unsigned)h) << 16);
}

// ---------------------------------------------------------------------------
// One-time: fp32 -> bf16 weight packing, W_ih|W_hh concat, embedding gather,
// bias fusion, state init.
// ---------------------------------------------------------------------------
__global__ void k_convert(const float* __restrict__ hidden_e,
                          const int*   __restrict__ target,
                          const float* __restrict__ W_l,
                          const float* __restrict__ W_sh,
                          const float* __restrict__ W_ih,
                          const float* __restrict__ b_ih,
                          const float* __restrict__ W_hh,
                          const float* __restrict__ b_hh,
                          const float* __restrict__ h0,
                          const float* __restrict__ c0,
                          const float* __restrict__ embedding,
                          unsigned short* __restrict__ hid_bf,
                          unsigned short* __restrict__ wl_bf,
                          unsigned short* __restrict__ wsh_bf,
                          unsigned short* __restrict__ wcat_bf,
                          unsigned short* __restrict__ emb_bf,
                          unsigned short* __restrict__ xh_bf,
                          float* __restrict__ bcat,
                          float* __restrict__ hbuf,
                          float* __restrict__ cbuf) {
    const size_t tid    = (size_t)blockIdx.x * blockDim.x + threadIdx.x;
    const size_t stride = (size_t)gridDim.x * blockDim.x;

    for (size_t i = tid; i < (size_t)BB*FF*N2; i += stride) hid_bf[i] = f2bf(hidden_e[i]);
    for (size_t i = tid; i < (size_t)NN*N2;    i += stride) wl_bf[i]  = f2bf(W_l[i]);
    for (size_t i = tid; i < (size_t)NN*N3;    i += stride) wsh_bf[i] = f2bf(W_sh[i]);
    for (size_t i = tid; i < (size_t)N4*K_IH;  i += stride) {
        size_t j = i / K_IH, k = i % K_IH;
        wcat_bf[j*K_CAT + k] = f2bf(W_ih[i]);
    }
    for (size_t i = tid; i < (size_t)N4*NN;    i += stride) {
        size_t j = i / NN, k = i % NN;
        wcat_bf[j*K_CAT + K_IH + k] = f2bf(W_hh[i]);
    }
    for (size_t i = tid; i < (size_t)BB*LL*EE; i += stride) {
        size_t bl = i / EE, e = i % EE;
        int word = target[bl];
        emb_bf[i] = f2bf(embedding[(size_t)word*EE + e]);
    }
    for (size_t i = tid; i < (size_t)N4; i += stride) bcat[i] = b_ih[i] + b_hh[i];
    for (size_t i = tid; i < (size_t)BB*NN; i += stride) {
        hbuf[i] = h0[i];
        cbuf[i] = c0[i];
        size_t b = i >> 11, n = i & (NN - 1);
        xh_bf[b*K_CAT + K_IH + n] = f2bf(h0[i]);   // h part of x|h row for step 0
    }
}

// ---------------------------------------------------------------------------
// bf16 WMMA GEMM:  C[M,N] = A[M,K] * B[N,K]^T + bias[N]   (all row-major)
// Each wave: one 16-row x (16*NT)-col tile.
// K loop: outer step 128 (all K here are multiples of 128), inner 4x32 fully
// unrolled with immediate load offsets. Within each 32-K sub-iteration we
// load A + all NT B fragments into distinct values FIRST (one long load
// clause, deep memory-level parallelism), then issue the NT WMMAs
// back-to-back; WMMAs 2..NT share the A operand with their predecessor, so
// the ISA reuse-A hint is set on them (legal: identical to previous instr).
// A fragment per ISA 16-bit 16x32 layout: lane m<16 -> K {base..base+7,
// 16+base..16+base+7} with base = 8*(lane>=16); B symmetric (lane = out col).
// Prefetch is unconditional: all operands sit in the middle of one contiguous
// workspace allocation, so a tail prefetch stays in mapped memory.
// ---------------------------------------------------------------------------
template <int NT>
__global__ void __launch_bounds__(256)
k_gemm_bf16(const unsigned short* __restrict__ A, int lda,
            const unsigned short* __restrict__ Bm, int ldb,
            const float* __restrict__ bias,
            float* __restrict__ C, int ldc,
            int M, int N, int K) {
    const int lane  = threadIdx.x & 31;
    const int wid   = (int)((blockIdx.x * blockDim.x + threadIdx.x) >> 5);
    const int tilesN = N / (16 * NT);
    const int tm = wid / tilesN;
    const int tn = wid % tilesN;
    const int m0 = tm * 16;
    if (m0 >= M) return;
    const int n0 = tn * 16 * NT;

    const int rc   = lane & 15;          // row (A) / out-col (B) within tile
    const int rsel = (lane >> 4) << 3;   // 0 or 8: K sub-block select

    v8f acc[NT] = {};

    const unsigned short* pa = A + (size_t)(m0 + rc) * lda + rsel;
    const unsigned short* pb[NT];
#pragma unroll
    for (int t = 0; t < NT; ++t)
        pb[t] = Bm + (size_t)(n0 + 16 * t + rc) * ldb + rsel;

    for (int k0 = 0; k0 < K; k0 += 128) {
        // pull the 256B/lane chunk 4 iterations ahead toward the caches (branchless)
        __builtin_prefetch(pa + k0 + 512, 0, 0);
#pragma unroll
        for (int t = 0; t < NT; ++t) __builtin_prefetch(pb[t] + k0 + 512, 0, 0);

#pragma unroll
        for (int kk = 0; kk < 128; kk += 32) {
            // ---- load phase: A + all B tiles as independent values ----
            FragAB a, b[NT];
            a.u[0] = *(const v8u*)(pa + k0 + kk);
            a.u[1] = *(const v8u*)(pa + k0 + kk + 16);
#pragma unroll
            for (int t = 0; t < NT; ++t) {
                b[t].u[0] = *(const v8u*)(pb[t] + k0 + kk);
                b[t].u[1] = *(const v8u*)(pb[t] + k0 + kk + 16);
            }
            // ---- compute phase: NT back-to-back WMMAs sharing A ----
            // (reuse_a must be a frontend constant: peel t=0, rest use true)
            acc[0] = __builtin_amdgcn_wmma_f32_16x16x32_bf16(
                         false, a.bf, false, b[0].bf, (short)0, acc[0],
                         false, false);
#pragma unroll
            for (int t = 1; t < NT; ++t) {
                acc[t] = __builtin_amdgcn_wmma_f32_16x16x32_bf16(
                             false, a.bf, false, b[t].bf, (short)0, acc[t],
                             true, false);
            }
        }
    }

#pragma unroll
    for (int t = 0; t < NT; ++t) {
#pragma unroll
        for (int r = 0; r < 8; ++r) {
            int row = m0 + rsel + r;           // VGPR r: M=r (lanes 0-15) / r+8
            int col = n0 + 16 * t + rc;
            C[(size_t)row * ldc + col] = acc[t][r] + bias[col];
        }
    }
}

// ---------------------------------------------------------------------------
// Attention: scores = (h*w_c) . proj[b,f,:], softmax over F, ctx = a . hidden_e
// Writes ctx (fp32) and the [emb_t | ctx] portion of the bf16 x|h row.
// One block per batch element.
// ---------------------------------------------------------------------------
__global__ void __launch_bounds__(256)
k_attend(const float* __restrict__ hbuf,
         const float* __restrict__ w_c,
         const float* __restrict__ proj,
         const unsigned short* __restrict__ hid_bf,
         const unsigned short* __restrict__ emb_bf,
         float* __restrict__ ctx,
         unsigned short* __restrict__ xh_bf,
         int t) {
    const int b    = blockIdx.x;
    const int lane = threadIdx.x & 31;
    const int wid  = threadIdx.x >> 5;

    __shared__ float sc[FF];
    __shared__ float av[FF];

    // scores: 8 waves x 8 features each, lane-strided dot of length N
    for (int fi = 0; fi < 8; ++fi) {
        const int f = wid * 8 + fi;
        const float* pr = proj + ((size_t)(b * FF + f)) * NN;
        const float* hp = hbuf + (size_t)b * NN;
        float s = 0.f;
        for (int n = lane; n < NN; n += 32) s += hp[n] * w_c[n] * pr[n];
        for (int off = 16; off > 0; off >>= 1) s += __shfl_xor(s, off, 32);
        if (lane == 0) sc[f] = s;
    }
    __syncthreads();

    // softmax over F=64 (redundant per-thread scan; tiny)
    float mx = -1e30f;
    for (int f = 0; f < FF; ++f) mx = fmaxf(mx, sc[f]);
    float sum = 0.f;
    for (int f = 0; f < FF; ++f) sum += __expf(sc[f] - mx);
    const float inv = 1.f / sum;
    if (threadIdx.x < FF) av[threadIdx.x] = __expf(sc[threadIdx.x] - mx) * inv;
    __syncthreads();

    // ctx[b,d] = sum_f a[f] * hidden_e[b,f,d]
    const unsigned short* hp = hid_bf + (size_t)b * FF * N2;
    for (int d = threadIdx.x; d < N2; d += 256) {
        float s = 0.f;
#pragma unroll 4
        for (int f = 0; f < FF; ++f) s += av[f] * bf2f(hp[(size_t)f * N2 + d]);
        ctx[(size_t)b * N2 + d] = s;
        xh_bf[(size_t)b * K_CAT + EE + d] = f2bf(s);
    }
    // copy teacher-forced embedding for this step
    const unsigned short* ep = emb_bf + ((size_t)b * LL + t) * EE;
    for (int e = threadIdx.x; e < EE; e += 256)
        xh_bf[(size_t)b * K_CAT + e] = ep[e];
}

// ---------------------------------------------------------------------------
// LSTM pointwise + build y = w_o*[h_new|ctx] + b_o (bf16) for logits GEMM.
// ---------------------------------------------------------------------------
__global__ void __launch_bounds__(256)
k_pointwise(const float* __restrict__ gates,
            float* __restrict__ hbuf,
            float* __restrict__ cbuf,
            unsigned short* __restrict__ xh_bf,
            const float* __restrict__ ctx,
            const float* __restrict__ w_o,
            const float* __restrict__ b_o,
            unsigned short* __restrict__ y_bf) {
    const int idx = blockIdx.x * 256 + threadIdx.x;   // grid covers B*2N

    if (idx < BB * NN) {
        const int b = idx >> 11, n = idx & (NN - 1);
        const float* g = gates + (size_t)b * N4;
        const float ig = 1.f / (1.f + __expf(-g[n]));
        const float fg = 1.f / (1.f + __expf(-g[NN + n]));
        const float gg = tanhf(g[2 * NN + n]);
        const float og = 1.f / (1.f + __expf(-g[3 * NN + n]));
        const float cn = fg * cbuf[idx] + ig * gg;
        const float hn = og * tanhf(cn);
        cbuf[idx] = cn;
        hbuf[idx] = hn;
        xh_bf[(size_t)b * K_CAT + K_IH + n] = f2bf(hn);          // h for next step
        y_bf[(size_t)b * N3 + n] =
            f2bf(w_o[(size_t)b * N3 + n] * hn + b_o[(size_t)b * N3 + n]);
    }
    if (idx < BB * N2) {
        const int b = idx >> 12, d = idx & (N2 - 1);
        const float cv = ctx[idx];
        y_bf[(size_t)b * N3 + NN + d] =
            f2bf(w_o[(size_t)b * N3 + NN + d] * cv + b_o[(size_t)b * N3 + NN + d]);
    }
}

// ---------------------------------------------------------------------------
// Row softmax over N=2048 -> out[b, t, :]
// ---------------------------------------------------------------------------
__global__ void __launch_bounds__(256)
k_softmax(const float* __restrict__ logits, float* __restrict__ out, int t) {
    const int b    = blockIdx.x;
    const int lane = threadIdx.x & 31;
    const int wid  = threadIdx.x >> 5;
    const float* lg = logits + (size_t)b * NN;
    __shared__ float redm[8], reds[8];

    float mx = -1e30f;
    for (int n = threadIdx.x; n < NN; n += 256) mx = fmaxf(mx, lg[n]);
    for (int off = 16; off > 0; off >>= 1) mx = fmaxf(mx, __shfl_xor(mx, off, 32));
    if (lane == 0) redm[wid] = mx;
    __syncthreads();
    mx = redm[0];
    for (int w = 1; w < 8; ++w) mx = fmaxf(mx, redm[w]);

    float sum = 0.f;
    for (int n = threadIdx.x; n < NN; n += 256) sum += __expf(lg[n] - mx);
    for (int off = 16; off > 0; off >>= 1) sum += __shfl_xor(sum, off, 32);
    if (lane == 0) reds[wid] = sum;
    __syncthreads();
    sum = 0.f;
    for (int w = 0; w < 8; ++w) sum += reds[w];
    const float inv = 1.f / sum;

    float* op = out + ((size_t)b * LL + t) * NN;
    for (int n = threadIdx.x; n < NN; n += 256) op[n] = __expf(lg[n] - mx) * inv;
}

// ---------------------------------------------------------------------------
extern "C" void kernel_launch(void* const* d_in, const int* in_sizes, int n_in,
                              void* d_out, int out_size, void* d_ws, size_t ws_size,
                              hipStream_t stream) {
    const float* hidden_e  = (const float*)d_in[0];
    const int*   target    = (const int*)  d_in[1];
    const float* W_l       = (const float*)d_in[2];
    const float* b_l       = (const float*)d_in[3];
    const float* W_sh      = (const float*)d_in[4];
    const float* b_sh      = (const float*)d_in[5];
    const float* W_ih      = (const float*)d_in[6];
    const float* b_ih      = (const float*)d_in[7];
    const float* W_hh      = (const float*)d_in[8];
    const float* b_hh      = (const float*)d_in[9];
    const float* b_o       = (const float*)d_in[10];
    const float* w_o       = (const float*)d_in[11];
    const float* w_c       = (const float*)d_in[12];
    const float* h0        = (const float*)d_in[13];
    const float* c0        = (const float*)d_in[14];
    const float* embedding = (const float*)d_in[15];
    float* out = (float*)d_out;

    // Workspace carve-out (~189 MB total).
    char* base = (char*)d_ws;
    size_t off = 0;
    auto wsa = [&](size_t bytes) -> void* {
        void* p = (void*)(base + off);
        off += (bytes + 255) & ~(size_t)255;
        return p;
    };
    unsigned short* hid_bf  = (unsigned short*)wsa((size_t)BB*FF*N2 * 2);
    unsigned short* wl_bf   = (unsigned short*)wsa((size_t)NN*N2    * 2);
    unsigned short* wsh_bf  = (unsigned short*)wsa((size_t)NN*N3    * 2);
    unsigned short* wcat_bf = (unsigned short*)wsa((size_t)N4*K_CAT * 2);
    unsigned short* emb_bf  = (unsigned short*)wsa((size_t)BB*LL*EE * 2);
    unsigned short* xh_bf   = (unsigned short*)wsa((size_t)BB*K_CAT * 2);
    unsigned short* y_bf    = (unsigned short*)wsa((size_t)BB*N3    * 2);
    float* proj   = (float*)wsa((size_t)BB*FF*NN * 4);
    float* gates  = (float*)wsa((size_t)BB*N4    * 4);
    float* logits = (float*)wsa((size_t)BB*NN    * 4);
    float* hbuf   = (float*)wsa((size_t)BB*NN    * 4);
    float* cbuf   = (float*)wsa((size_t)BB*NN    * 4);
    float* ctxbuf = (float*)wsa((size_t)BB*N2    * 4);
    float* bcat   = (float*)wsa((size_t)N4       * 4);

    // 1) one-time pack/convert/init
    k_convert<<<2048, 256, 0, stream>>>(hidden_e, target, W_l, W_sh, W_ih, b_ih,
                                        W_hh, b_hh, h0, c0, embedding,
                                        hid_bf, wl_bf, wsh_bf, wcat_bf, emb_bf,
                                        xh_bf, bcat, hbuf, cbuf);

    // 2) proj = hidden_e @ W_l^T + b_l : [2048 x 4096] x [2048 x 4096]^T
    //    waves = (2048/16)*(2048/64) = 4096 -> 512 blocks of 8 waves
    k_gemm_bf16<4><<<512, 256, 0, stream>>>(hid_bf, N2, wl_bf, N2, b_l,
                                            proj, NN, BB*FF, NN, N2);

    // 3) sequential decode
    for (int t = 0; t < LL; ++t) {
        k_attend<<<BB, 256, 0, stream>>>(hbuf, w_c, proj, hid_bf, emb_bf,
                                         ctxbuf, xh_bf, t);
        // gates = [emb|ctx|h] @ [W_ih|W_hh]^T + (b_ih+b_hh)
        // waves = (32/16)*(8192/64) = 256 -> 32 blocks
        k_gemm_bf16<4><<<32, 256, 0, stream>>>(xh_bf, K_CAT, wcat_bf, K_CAT, bcat,
                                               gates, N4, BB, N4, K_CAT);
        k_pointwise<<<(BB*N2 + 255)/256, 256, 0, stream>>>(gates, hbuf, cbuf, xh_bf,
                                                           ctxbuf, w_o, b_o, y_bf);
        // logits = y @ W_sh^T + b_sh : waves = 2*32 = 64 -> 8 blocks
        k_gemm_bf16<4><<<8, 256, 0, stream>>>(y_bf, N3, wsh_bf, N3, b_sh,
                                              logits, NN, BB, NN, N3);
        k_softmax<<<BB, 256, 0, stream>>>(logits, out, t);
    }
}
